// RNNRobotModel_29729763623471
// MI455X (gfx1250) — matrix operand order repaired and checked by reading, not compiled, not run on hardware
//
#include <hip/hip_runtime.h>
#include <hip/hip_bf16.h>

typedef _Float16 h8  __attribute__((ext_vector_type(8)));
typedef _Float16 v16h __attribute__((ext_vector_type(16)));
typedef float    v8f  __attribute__((ext_vector_type(8)));

#define B_TOT 8192
#define T_LEN 512
#define HDIM  64
#define WPB   2          // waves per block -> 256 workgroups across the device
#define SH    104        // h-buffer stride (halves), 16B aligned, conflict-free
#define SA    72         // a1-buffer stride
#define SWH   72         // W_hh stride
#define SWI   40         // W_ih (padded to 32 K cols) stride
#define SW1   72
#define SW2   72
#define DT    0.016f

union frag_u { v16h v; h8 h[2]; };

__device__ __forceinline__ v16h ld_afrag(const _Float16* hb, int lane, int c0) {
    // A 16x32 f16 layout: lane%16 = row, lane/16 selects K-subgroup of 8
    const _Float16* p = hb + (lane & 15) * SH + c0 + ((lane >> 4) << 3);
    frag_u u;
    u.h[0] = *(const h8*)(p);
    u.h[1] = *(const h8*)(p + 16);
    return u.v;
}

__device__ __forceinline__ v16h ld_bfrag(const _Float16* W, int stride, int n, int kbase, int lane) {
    // B 32x16 f16 layout: lane%16 = column n, lane/16 selects K-subgroup of 8
    const _Float16* p = W + n * stride + kbase + ((lane >> 4) << 3);
    frag_u u;
    u.h[0] = *(const h8*)(p);
    u.h[1] = *(const h8*)(p + 16);
    return u.v;
}

__device__ __forceinline__ v8f wmma_f16(v16h a, v16h b, v8f c) {
    return __builtin_amdgcn_wmma_f32_16x16x32_f16(false, a, false, b, (short)0, c, false, false);
}

// fast nonlinearities: single v_exp_f32 + single v_rcp_f32 (no IEEE div expansion)
__device__ __forceinline__ float fsigmoid(float x) {
    return __builtin_amdgcn_rcpf(1.0f + __expf(-x));
}
__device__ __forceinline__ float ftanh(float x) {
    return 1.0f - 2.0f * __builtin_amdgcn_rcpf(__expf(2.0f * x) + 1.0f);
}

__global__ __launch_bounds__(WPB * 32, 1)
void rnn_robot_fused(const float* __restrict__ init_state,
                     const float* __restrict__ cmd,
                     const float* __restrict__ W_ih, const float* __restrict__ W_hh,
                     const float* __restrict__ b_ih, const float* __restrict__ b_hh,
                     const float* __restrict__ W1,   const float* __restrict__ b1,
                     const float* __restrict__ W2,   const float* __restrict__ b2,
                     float* __restrict__ out) {
    __shared__ _Float16 sWh[192 * SWH];
    __shared__ _Float16 sWi[192 * SWI];
    __shared__ _Float16 sW1[HDIM * SW1];
    __shared__ _Float16 sW2[16 * SW2];
    __shared__ float    sbih[192], sbhh[192], sb1[HDIM], sb2[4];
    __shared__ _Float16 sH[WPB * 16 * SH];
    __shared__ _Float16 sA[WPB * 16 * SA];
    __shared__ float    sAcc[WPB * 16 * 4];

    const int tid  = threadIdx.x;
    const int lane = tid & 31;
    const int wave = tid >> 5;
    const int nthr = WPB * 32;

    // ---- stage weights to LDS as f16 (row-major [N][K], zero-padded) ----
    for (int i = tid; i < 192 * SWH; i += nthr) {
        int r = i / SWH, c = i % SWH;
        sWh[i] = (c < HDIM) ? (_Float16)W_hh[r * HDIM + c] : (_Float16)0.0f;
    }
    for (int i = tid; i < 192 * SWI; i += nthr) {
        int r = i / SWI, c = i % SWI;
        sWi[i] = (c < 6) ? (_Float16)W_ih[r * 6 + c] : (_Float16)0.0f;
    }
    for (int i = tid; i < HDIM * SW1; i += nthr) {
        int r = i / SW1, c = i % SW1;
        sW1[i] = (c < HDIM) ? (_Float16)W1[r * HDIM + c] : (_Float16)0.0f;
    }
    for (int i = tid; i < 16 * SW2; i += nthr) {
        int r = i / SW2, c = i % SW2;
        sW2[i] = (r < 3 && c < HDIM) ? (_Float16)W2[r * HDIM + c] : (_Float16)0.0f;
    }
    for (int i = tid; i < 192; i += nthr) { sbih[i] = b_ih[i]; sbhh[i] = b_hh[i]; }
    for (int i = tid; i < HDIM; i += nthr) sb1[i] = b1[i];
    if (tid < 4) sb2[tid] = (tid < 3) ? b2[tid] : 0.0f;
    for (int i = tid; i < WPB * 16 * SH; i += nthr) sH[i] = (_Float16)0.0f;   // h0 = 0, pads = 0
    for (int i = tid; i < WPB * 16 * SA; i += nthr) sA[i] = (_Float16)0.0f;
    __syncthreads();

    _Float16* hb   = sH   + wave * 16 * SH;
    _Float16* ab   = sA   + wave * 16 * SA;
    float*    accb = sAcc + wave * 16 * 4;

    const int m = lane;                                  // physics row (lanes 0..15)
    const int brow = blockIdx.x * (WPB * 16) + wave * 16 + m;

    float px = 0.f, py = 0.f, th = 0.f, vxw = 0.f, vyw = 0.f, om = 0.f, vxb = 0.f, vyb = 0.f;
    if (lane < 16) {
        const float* s = init_state + (size_t)brow * 6;
        px = s[0]; py = s[1]; th = s[2]; vxw = s[3]; vyw = s[4]; om = s[5];
        float2* o0 = (float2*)(out + (size_t)brow * (T_LEN + 1) * 6);
        o0[0] = make_float2(px, py); o0[1] = make_float2(th, vxw); o0[2] = make_float2(vyw, om);
        float cc = __cosf(th), ss = __sinf(th);
        vxb = vxw * cc + vyw * ss;
        vyb = -vxw * ss + vyw * cc;
        hb[m * SH + 64] = (_Float16)vxb;
        hb[m * SH + 65] = (_Float16)vyb;
        hb[m * SH + 66] = (_Float16)om;
    }

    const int grpA = (lane >> 4);   // K-subgroup select
    const int ncol = lane & 15;     // B/C/D column within tile

    // hidden state kept in registers: hreg[tt][v] = h[v + 8*grpA][tt*16 + ncol]
    // (exactly the elements this lane produces in the D tile each step)
    float hreg[4][8];
#pragma unroll
    for (int tt = 0; tt < 4; ++tt)
#pragma unroll
        for (int v = 0; v < 8; ++v) hreg[tt][v] = 0.0f;

#pragma unroll 1
    for (int t = 0; t < T_LEN; ++t) {
        // ---- stream this step's command into x-columns of h buffer ----
        if (lane < 16) {
            const float* cp = cmd + ((size_t)brow * T_LEN + t) * 3;
            float c0 = cp[0], c1 = cp[1], c2 = cp[2];
            hb[m * SH + 67] = (_Float16)c0;
            hb[m * SH + 68] = (_Float16)c1;
            hb[m * SH + 69] = (_Float16)c2;
            if (t + 1 < T_LEN) __builtin_prefetch(cp + 3, 0, 1);
        }

        // A fragments: K-blocks 0 (h 0..31), 1 (h 32..63), 2 (x, cols 64..95)
        v16h aK0 = ld_afrag(hb, lane, 0);
        v16h aK1 = ld_afrag(hb, lane, 32);
        v16h aK2 = ld_afrag(hb, lane, 64);

        // ---- GRU: process 4 column tiles of the 64-wide gates ----
#pragma unroll
        for (int tt = 0; tt < 4; ++tt) {
            const int nr = tt * 16 + ncol;        // r-gate weight row
            const int nz = 64 + nr;               // z-gate
            const int nn = 128 + nr;              // n-gate
            v8f cr, cz, cin, chn;
            const float br = sbih[nr] + sbhh[nr];
            const float bz = sbih[nz] + sbhh[nz];
            const float bi = sbih[nn];
            const float bh = sbhh[nn];
#pragma unroll
            for (int v = 0; v < 8; ++v) { cr[v] = br; cz[v] = bz; cin[v] = bi; chn[v] = bh; }

            cr  = wmma_f16(aK0, ld_bfrag(sWh, SWH, nr, 0,  lane), cr);
            cr  = wmma_f16(aK1, ld_bfrag(sWh, SWH, nr, 32, lane), cr);
            cr  = wmma_f16(aK2, ld_bfrag(sWi, SWI, nr, 0,  lane), cr);   // x @ W_ih_r
            cz  = wmma_f16(aK0, ld_bfrag(sWh, SWH, nz, 0,  lane), cz);
            cz  = wmma_f16(aK1, ld_bfrag(sWh, SWH, nz, 32, lane), cz);
            cz  = wmma_f16(aK2, ld_bfrag(sWi, SWI, nz, 0,  lane), cz);
            chn = wmma_f16(aK0, ld_bfrag(sWh, SWH, nn, 0,  lane), chn);  // h_n (gets r*)
            chn = wmma_f16(aK1, ld_bfrag(sWh, SWH, nn, 32, lane), chn);
            cin = wmma_f16(aK2, ld_bfrag(sWi, SWI, nn, 0,  lane), cin);  // i_n separate

#pragma unroll
            for (int v = 0; v < 8; ++v) {
                const int mm = v + (grpA << 3);
                float r  = fsigmoid(cr[v]);
                float z  = fsigmoid(cz[v]);
                float nl = ftanh(cin[v] + r * chn[v]);
                float hn = (1.0f - z) * nl + z * hreg[tt][v];   // h_old from registers
                hreg[tt][v] = hn;
                hb[mm * SH + nr] = (_Float16)hn;
            }
        }

        // ---- MLP layer 1: a1 = tanh(h_new @ W1^T + b1) ----
        v16h hA0 = ld_afrag(hb, lane, 0);
        v16h hA1 = ld_afrag(hb, lane, 32);
#pragma unroll
        for (int tt = 0; tt < 4; ++tt) {
            const int n1 = tt * 16 + ncol;
            v8f c1;
            const float bb = sb1[n1];
#pragma unroll
            for (int v = 0; v < 8; ++v) c1[v] = bb;
            c1 = wmma_f16(hA0, ld_bfrag(sW1, SW1, n1, 0,  lane), c1);
            c1 = wmma_f16(hA1, ld_bfrag(sW1, SW1, n1, 32, lane), c1);
#pragma unroll
            for (int v = 0; v < 8; ++v) {
                const int mm = v + (grpA << 3);
                ab[mm * SA + n1] = (_Float16)ftanh(c1[v]);
            }
        }

        // ---- MLP layer 2: acc = a1 @ W2p^T + b2 (cols 0..2 valid) ----
        v16h mA0 = ld_afrag(ab, lane, 0);
        v16h mA1 = ld_afrag(ab, lane, 32);
        {
            v8f c2;
            const float bb = (ncol < 3) ? sb2[ncol] : 0.0f;
#pragma unroll
            for (int v = 0; v < 8; ++v) c2[v] = bb;
            c2 = wmma_f16(mA0, ld_bfrag(sW2, SW2, ncol, 0,  lane), c2);
            c2 = wmma_f16(mA1, ld_bfrag(sW2, SW2, ncol, 32, lane), c2);
            if (ncol < 3) {
#pragma unroll
                for (int v = 0; v < 8; ++v)
                    accb[(v + (grpA << 3)) * 4 + ncol] = c2[v];
            }
        }

        // ---- physics integration (lane m owns batch row) ----
        if (lane < 16) {
            float a0 = accb[m * 4 + 0];
            float a1 = accb[m * 4 + 1];
            float a2 = accb[m * 4 + 2];
            float nvxb = vxb + a0 * DT;
            float nvyb = vyb + a1 * DT;
            float nom  = om  + a2 * DT;
            float nth  = th + nom * DT;
            const float TWO_PI = 6.283185307179586f;
            nth = nth - TWO_PI * rintf(nth * (1.0f / TWO_PI));   // == atan2(sin,cos) wrap
            float cn = __cosf(nth), sn = __sinf(nth);
            float nvxw = nvxb * cn - nvyb * sn;
            float nvyw = nvxb * sn + nvxb * cn;   // reproduces reference expression
            px += nvxw * DT;
            py += nvyw * DT;
            th = nth; vxw = nvxw; vyw = nvyw; om = nom;
            // next step's body-frame velocities (reference recomputes from new state)
            vxb = vxw * cn + vyw * sn;
            vyb = -vxw * sn + vyw * cn;
            hb[m * SH + 64] = (_Float16)vxb;
            hb[m * SH + 65] = (_Float16)vyb;
            hb[m * SH + 66] = (_Float16)om;
            float2* o = (float2*)(out + ((size_t)brow * (T_LEN + 1) + (t + 1)) * 6);
            o[0] = make_float2(px, py);
            o[1] = make_float2(th, vxw);
            o[2] = make_float2(vyw, om);
        }
    }
}

extern "C" void kernel_launch(void* const* d_in, const int* in_sizes, int n_in,
                              void* d_out, int out_size, void* d_ws, size_t ws_size,
                              hipStream_t stream) {
    const float* init_state = (const float*)d_in[0];
    const float* cmd        = (const float*)d_in[1];
    const float* W_ih       = (const float*)d_in[2];
    const float* W_hh       = (const float*)d_in[3];
    const float* b_ih       = (const float*)d_in[4];
    const float* b_hh       = (const float*)d_in[5];
    const float* W1         = (const float*)d_in[6];
    const float* b1         = (const float*)d_in[7];
    const float* W2         = (const float*)d_in[8];
    const float* b2         = (const float*)d_in[9];
    float* out = (float*)d_out;

    dim3 grid(B_TOT / (WPB * 16));   // 256 blocks
    dim3 block(WPB * 32);            // 2 wave32 per block
    rnn_robot_fused<<<grid, block, 0, stream>>>(init_state, cmd, W_ih, W_hh,
                                                b_ih, b_hh, W1, b1, W2, b2, out);
}
